// LSTMAutoencoder_53369263620301
// MI455X (gfx1250) — matrix operand order repaired and checked by reading, not compiled, not run on hardware
//
#include <hip/hip_runtime.h>

#define HID 32
#define WIN 120
#define NB  16384

typedef _Float16 v16h __attribute__((ext_vector_type(16)));
typedef float    v8f  __attribute__((ext_vector_type(8)));
typedef float    v2f  __attribute__((ext_vector_type(2)));

__device__ __forceinline__ float fexp(float x){ return __builtin_amdgcn_exp2f(x*1.4426950408889634f); }
__device__ __forceinline__ float sigm(float x){ return __builtin_amdgcn_rcpf(1.0f + fexp(-x)); }

#if __has_builtin(__builtin_amdgcn_tanhf)
__device__ __forceinline__ float tanhx(float x){ return __builtin_amdgcn_tanhf(x); }
#elif __has_builtin(__builtin_amdgcn_tanh_f32)
__device__ __forceinline__ float tanhx(float x){ return __builtin_amdgcn_tanh_f32(x); }
#else
__device__ __forceinline__ float tanhx(float x){ return 1.0f - 2.0f*__builtin_amdgcn_rcpf(1.0f + fexp(2.0f*x)); }
#endif

__device__ __forceinline__ v8f wmma16(v16h a, v16h b, v8f c){
  return __builtin_amdgcn_wmma_f32_16x16x32_f16(false, a, false, b, (short)0, c, false, false);
}

#if __has_builtin(__builtin_amdgcn_wmma_f32_16x16x4_f32)
#define HAS_WMMA_F32K4 1
__device__ __forceinline__ v8f wmma4(v2f a, v2f b, v8f c){
  return __builtin_amdgcn_wmma_f32_16x16x4_f32(false, a, false, b, (short)0, c, false, false);
}
#else
#define HAS_WMMA_F32K4 0
#endif

// Build f16 A-layout tile t (rows 16t..16t+15, K=32) from row-major W[128][32].
// Lane L (m = L&15): lanes<16 hold K=0..15, lanes>=16 hold K=16..31.
__device__ __forceinline__ v16h build_A(const float* __restrict__ W, int t, int lane){
  const int m  = t*16 + (lane & 15);
  const int k0 = (lane & 16) ? 16 : 0;
  const float* row = W + m*HID;
  v16h a;
  #pragma unroll
  for(int q=0;q<8;q++){
    a[q]   = (_Float16)row[k0+q];
    a[8+q] = (_Float16)row[k0+8+q];
  }
  return a;
}

// Gate tiles in C layout: tile t, VGPR r, lane L -> gate g = 16t + r + 8*(L/16), batch n = L&15.
// i: tiles 0,1  f: 2,3  g: 4,5  o: 6,7. Hidden j = r + 8*hi (+16 for odd tile).
__device__ __forceinline__ void lstm_elem(const v8f d[8], v8f& c0, v8f& c1,
                                          float* __restrict__ h0, float* __restrict__ h1){
  #pragma unroll
  for(int r=0;r<8;r++){
    float i0 = sigm(d[0][r]), f0 = sigm(d[2][r]), g0 = tanhx(d[4][r]), o0 = sigm(d[6][r]);
    float cc0 = f0*c0[r] + i0*g0; c0[r] = cc0; h0[r] = o0*tanhx(cc0);
    float i1 = sigm(d[1][r]), f1 = sigm(d[3][r]), g1 = tanhx(d[5][r]), o1 = sigm(d[7][r]);
    float cc1 = f1*c1[r] + i1*g1; c1[r] = cc1; h1[r] = o1*tanhx(cc1);
  }
}

// Convert new h from C layout (h0[r]: j=r+8*hi ; h1[r]: j=16+r+8*hi) to f16 B layout
// (lane<16 holds K=0..15, lane>=16 holds K=16..31, column n=L&15) via lane<->lane+16 swap.
__device__ __forceinline__ v16h cvt_CtoB(const float* __restrict__ h0, const float* __restrict__ h1, bool lol){
  v16h b;
  #pragma unroll
  for(int r=0;r<8;r++){
    float s0 = __shfl_xor(h0[r], 16, 32);
    float s1 = __shfl_xor(h1[r], 16, 32);
    b[r]   = (_Float16)(lol ? h0[r] : s1);
    b[8+r] = (_Float16)(lol ? s0 : h1[r]);
  }
  return b;
}

__global__ __launch_bounds__(128) void lstm_ae_kernel(
    const float* __restrict__ x,
    const float* __restrict__ eWih, const float* __restrict__ eWhh,
    const float* __restrict__ ebih, const float* __restrict__ ebhh,
    const float* __restrict__ elW,  const float* __restrict__ elb,
    const float* __restrict__ dlW,  const float* __restrict__ dlb,
    const float* __restrict__ dWih, const float* __restrict__ dWhh,
    const float* __restrict__ dbih, const float* __restrict__ dbhh,
    const float* __restrict__ oW,   const float* __restrict__ ob,
    float* __restrict__ out)
{
  __shared__ float s_xy[4][16*WIN];     // x buffer, later y buffer (per wave)
  __shared__ float s_hT[4][16][HID];
  __shared__ float s_d0[4][16][HID];

  const int  tid  = threadIdx.x;
  const int  wave = tid >> 5;
  const int  lane = tid & 31;
  const int  n    = lane & 15;
  const bool lol  = (lane & 16) == 0;
  const int  hi8  = lol ? 0 : 8;
  const long base = (long)blockIdx.x*64 + wave*16;

  // ---- stage x (16 samples x 120, contiguous) into LDS, coalesced ----
  {
    const float* src = x + base*WIN;
    float* dst = s_xy[wave];
    for(int i=lane; i<16*WIN; i+=32) dst[i] = src[i];
  }
  __syncthreads();

  // ---- encoder weights resident in VGPRs ----
  v16h Ae[8];
  #pragma unroll
  for(int t=0;t<8;t++) Ae[t] = build_A(eWhh, t, lane);

#if HAS_WMMA_F32K4
  v2f Ag[8];
  #pragma unroll
  for(int t=0;t<8;t++){
    int g = t*16 + n;
    Ag[t].x = lol ? eWih[g] : 0.0f;
    Ag[t].y = lol ? (ebih[g] + ebhh[g]) : 0.0f;
  }
#else
  v16h Ag[8];
  #pragma unroll
  for(int t=0;t<8;t++){
    int g = t*16 + n;
    #pragma unroll
    for(int q=0;q<16;q++) Ag[t][q] = (_Float16)0.0f;
    if(lol){ Ag[t][0] = (_Float16)eWih[g]; Ag[t][1] = (_Float16)(ebih[g]+ebhh[g]); }
  }
#endif

  v8f c0, c1, zero8;
  float h0[8], h1[8];
  v16h hB;
  #pragma unroll
  for(int r=0;r<8;r++){ c0[r]=0.0f; c1[r]=0.0f; zero8[r]=0.0f; }
  #pragma unroll
  for(int q=0;q<16;q++) hB[q] = (_Float16)0.0f;

  // ---- encoder: 120 sequential steps, 16 batch per wave ----
  // Software-pipeline the per-step x fetch so the DS wait is hidden behind the
  // elementwise VALU phase of the previous step.
  float xv = s_xy[wave][n*WIN + 0];
  for(int ts=0; ts<WIN; ++ts){
    v8f d[8];
#if HAS_WMMA_F32K4
    v2f uB; uB.x = lol ? xv : 0.0f; uB.y = lol ? 1.0f : 0.0f;
    #pragma unroll
    for(int t=0;t<8;t++){
      v8f cx = wmma4(Ag[t], uB, zero8);      // x*Wih + bias (f32, K=4)
      d[t]   = wmma16(Ae[t], hB, cx);        // + Whh·h^T   (f16, K=32)
    }
#else
    v16h uB;
    #pragma unroll
    for(int q=0;q<16;q++) uB[q] = (_Float16)0.0f;
    if(lol){ uB[0] = (_Float16)xv; uB[1] = (_Float16)1.0f; }
    #pragma unroll
    for(int t=0;t<8;t++){
      v8f cx = wmma16(Ag[t], uB, zero8);
      d[t]   = wmma16(Ae[t], hB, cx);
    }
#endif
    // issue next step's x load early (covered by the activation math below)
    float xv_next = (ts+1 < WIN) ? s_xy[wave][n*WIN + ts + 1] : 0.0f;
    lstm_elem(d, c0, c1, h0, h1);
    hB = cvt_CtoB(h0, h1, lol);
    xv = xv_next;
  }

  // ---- hT -> LDS (C layout scatter) ----
  #pragma unroll
  for(int r=0;r<8;r++){
    s_hT[wave][n][hi8+r]     = h0[r];
    s_hT[wave][n][16+hi8+r]  = h1[r];
  }
  __syncthreads();

  // ---- latent: z = hT·enclW^T + b (tiny, per-lane VALU) ----
  float zv[8];
  #pragma unroll
  for(int l=0;l<8;l++){
    float acc = elb[l];
    #pragma unroll
    for(int j=0;j<HID;j++) acc += elW[l*HID+j] * s_hT[wave][n][j];
    zv[l] = acc;
  }
  // d0 = z·declW^T + b : lanes split the 32 rows (lo: 0..15, hi: 16..31)
  {
    const int m0 = lol ? 0 : 16;
    #pragma unroll
    for(int k=0;k<16;k++){
      const int m = m0 + k;
      float acc = dlb[m];
      #pragma unroll
      for(int l=0;l<8;l++) acc += dlW[m*8+l]*zv[l];
      s_d0[wave][n][m] = acc;
    }
  }
  __syncthreads();

  // d0 in B layout (f16)
  v16h d0B;
  {
    const int m0 = lol ? 0 : 16;
    #pragma unroll
    for(int q=0;q<16;q++) d0B[q] = (_Float16)s_d0[wave][n][m0+q];
  }

  // ---- decoder weights + time-invariant input projection xp = Wih·d0^T + bias ----
  v16h Ad[8]; v8f xp[8];
  #pragma unroll
  for(int t=0;t<8;t++) Ad[t] = build_A(dWhh, t, lane);
  #pragma unroll
  for(int t=0;t<8;t++){
    v16h Aw = build_A(dWih, t, lane);
    v8f cb;
    const int g0 = t*16 + hi8;
    #pragma unroll
    for(int r=0;r<8;r++) cb[r] = dbih[g0+r] + dbhh[g0+r];
    xp[t] = wmma16(Aw, d0B, cb);
  }

  float ow0[8], ow1[8];
  #pragma unroll
  for(int r=0;r<8;r++){ ow0[r] = oW[hi8+r]; ow1[r] = oW[16+hi8+r]; }
  const float obv = ob[0];

  #pragma unroll
  for(int r=0;r<8;r++){ c0[r]=0.0f; c1[r]=0.0f; }
  #pragma unroll
  for(int q=0;q<16;q++) hB[q] = (_Float16)0.0f;

  // ---- decoder: 120 steps; xp reused as WMMA C input each step ----
  for(int ts=0; ts<WIN; ++ts){
    v8f d[8];
    #pragma unroll
    for(int t=0;t<8;t++) d[t] = wmma16(Ad[t], hB, xp[t]);
    lstm_elem(d, c0, c1, h0, h1);

    // y = out_W·h + out_b : per-lane partial over 16 hidden, pair-reduce across lane+16
    float part = 0.0f;
    #pragma unroll
    for(int r=0;r<8;r++) part += h0[r]*ow0[r] + h1[r]*ow1[r];
    part += __shfl_xor(part, 16, 32);
    if(lol) s_xy[wave][n*WIN + ts] = part + obv;

    hB = cvt_CtoB(h0, h1, lol);
  }
  __syncthreads();

  // ---- coalesced writeback of y ----
  {
    float* dst = out + base*WIN;
    const float* src = s_xy[wave];
    for(int i=lane; i<16*WIN; i+=32) dst[i] = src[i];
  }
}

extern "C" void kernel_launch(void* const* d_in, const int* in_sizes, int n_in,
                              void* d_out, int out_size, void* d_ws, size_t ws_size,
                              hipStream_t stream){
  const float* x    = (const float*)d_in[0];
  const float* eWih = (const float*)d_in[1];
  const float* eWhh = (const float*)d_in[2];
  const float* ebih = (const float*)d_in[3];
  const float* ebhh = (const float*)d_in[4];
  const float* elW  = (const float*)d_in[5];
  const float* elb  = (const float*)d_in[6];
  const float* dlW  = (const float*)d_in[7];
  const float* dlb  = (const float*)d_in[8];
  const float* dWih = (const float*)d_in[9];
  const float* dWhh = (const float*)d_in[10];
  const float* dbih = (const float*)d_in[11];
  const float* dbhh = (const float*)d_in[12];
  const float* oW   = (const float*)d_in[13];
  const float* obp  = (const float*)d_in[14];
  float* out = (float*)d_out;

  dim3 grid(NB/64);   // 64 samples per 128-thread (4-wave) workgroup
  dim3 block(128);
  lstm_ae_kernel<<<grid, block, 0, stream>>>(x, eWih, eWhh, ebih, ebhh,
                                             elW, elb, dlW, dlb,
                                             dWih, dWhh, dbih, dbhh,
                                             oW, obp, out);
}
